// CityAgglomerationGNN_5909874999694
// MI455X (gfx1250) — matrix operand order repaired and checked by reading, not compiled, not run on hardware
//
#include <hip/hip_runtime.h>
#include <stdint.h>

// ---------------------------------------------------------------------------
// CDNA5 (gfx1250) types for WMMA
// ---------------------------------------------------------------------------
typedef __attribute__((ext_vector_type(16))) __bf16       v16bf;
typedef __attribute__((ext_vector_type(8)))  float        v8f;
typedef __attribute__((ext_vector_type(4)))  unsigned int u32x4;

union BFrag { u32x4 q[2]; v16bf v; };

__device__ __forceinline__ unsigned short f32_to_bf16_rne(float f) {
  union { float f; unsigned int u; } cvt;
  cvt.f = f;
  unsigned int u = cvt.u;
  u += 0x7FFFu + ((u >> 16) & 1u);   // round-to-nearest-even
  return (unsigned short)(u >> 16);
}

// ---------------------------------------------------------------------------
// fp32 -> bf16 bulk convert (float4 in, 4x bf16 packed u64 out)
// ---------------------------------------------------------------------------
__global__ void cvt_f32_bf16_kernel(const float* __restrict__ in,
                                    unsigned short* __restrict__ out, int n4) {
  const int i = blockIdx.x * blockDim.x + threadIdx.x;
  if (i >= n4) return;
  const float4 v = ((const float4*)in)[i];
  const unsigned long long p =
      (unsigned long long)f32_to_bf16_rne(v.x) |
      ((unsigned long long)f32_to_bf16_rne(v.y) << 16) |
      ((unsigned long long)f32_to_bf16_rne(v.z) << 32) |
      ((unsigned long long)f32_to_bf16_rne(v.w) << 48);
  ((unsigned long long*)out)[i] = p;
}

// W[K][Ncol] fp32 -> Wt[Ncol][K] bf16 (tiny, L2-resident)
__global__ void cvt_w_tr_kernel(const float* __restrict__ W,
                                unsigned short* __restrict__ Wt, int K, int Ncol) {
  const int i = blockIdx.x * blockDim.x + threadIdx.x;
  if (i >= K * Ncol) return;
  const int k = i % K, nn = i / K;
  Wt[i] = f32_to_bf16_rne(W[(size_t)k * Ncol + nn]);
}

// ---------------------------------------------------------------------------
// WMMA GEMM: H[M x NCOLS] = Xb[M x KDIM](bf16) * Wt[NCOLS x KDIM](bf16, pre-T)
// One block = 16-row slab. The whole 16 x KDIM bf16 A slab is staged to LDS
// once with GLOBAL_LOAD_ASYNC_TO_LDS_B128 (ASYNCcnt), then the K-loop is pure
// ds_load_b128 + v_wmma with no barriers. B fragments come straight from
// global (weights live in L2).
// ---------------------------------------------------------------------------
template<int KDIM, int NCOLS>
__global__ void __launch_bounds__((NCOLS / 16) * 32)
gemm_bf16_wmma(const unsigned short* __restrict__ Xb,
               const unsigned short* __restrict__ Wt,
               float* __restrict__ H, int M) {
  constexpr int NW = NCOLS / 16;
  constexpr int NT = NW * 32;
  __shared__ alignas(16) unsigned short sA[16 * KDIM];   // [row][k], stride KDIM

  const int tid  = threadIdx.x;
  const int lane = tid & 31;
  const int wave = tid >> 5;
  const int rowBase = blockIdx.x * 16;

  // ---- async-stage A slab: 16 rows x KDIM bf16 = 16*KDIM*2 bytes ----
  {
    constexpr int VPR  = (KDIM * 2) / 16;   // 16B vectors per row
    constexpr int NVEC = 16 * VPR;
    const unsigned long long gbase = (unsigned long long)(const void*)Xb;
    const unsigned ldsBase = (unsigned)(uintptr_t)(void*)sA;
    for (int i = tid; i < NVEC; i += NT) {
      int row = rowBase + (i / VPR);
      if (row >= M) row = M - 1;                       // defensive (M%16==0 here)
      const unsigned goff = (unsigned)row * (KDIM * 2) + (unsigned)(i % VPR) * 16;
      const unsigned loff = ldsBase + (unsigned)i * 16;
      asm volatile("global_load_async_to_lds_b128 %0, %1, %2 offset:0"
                   :: "v"(loff), "v"(goff), "s"(gbase)
                   : "memory");
    }
    asm volatile("s_wait_asynccnt 0" ::: "memory");
  }
  __syncthreads();

  v8f acc;
#pragma unroll
  for (int i = 0; i < 8; ++i) acc[i] = 0.0f;

  // Fragment coordinates (ISA 7.12.2 layouts).
  const int m  = lane & 15;
  const int ah = (lane >> 4) * 8;          // A: lane-half -> K+0..7 / K+8..15 (lo), +16 (hi)
  const int n  = wave * 16 + (lane & 15);
  const int bh = (lane >> 4) * 8;          // B: lane-half -> K+0..7 / K+8..15 (lo), +16 (hi)
  const unsigned short* wrow = Wt + (size_t)n * KDIM;

#pragma unroll
  for (int kk = 0; kk < KDIM; kk += 32) {
    BFrag afrag, bfrag;
    afrag.q[0] = *(const u32x4*)&sA[m * KDIM + kk + ah];
    afrag.q[1] = *(const u32x4*)&sA[m * KDIM + kk + ah + 16];
    bfrag.q[0] = *(const u32x4*)&wrow[kk + bh];
    bfrag.q[1] = *(const u32x4*)&wrow[kk + bh + 16];
    acc = __builtin_amdgcn_wmma_f32_16x16x32_bf16(
        false, afrag.v, false, bfrag.v, (short)0, acc, false, false);
  }

  // C/D layout: vgpr j -> row = j + 8*(lane>=16), col = lane%16.
  const int mh = (lane >> 4) * 8;
#pragma unroll
  for (int j = 0; j < 8; ++j) {
    const int row = rowBase + mh + j;
    if (row < M) H[(size_t)row * NCOLS + n] = acc[j];
  }
}

// ---------------------------------------------------------------------------
// Elementwise / graph kernels
// ---------------------------------------------------------------------------
__global__ void fill_kernel(float* __restrict__ p, float v, int n) {
  const int i = blockIdx.x * blockDim.x + threadIdx.x;
  if (i < n) p[i] = v;
}

__global__ void deg_accum_kernel(const int* __restrict__ dst,
                                 float* __restrict__ deg, int E) {
  const int e = blockIdx.x * blockDim.x + threadIdx.x;
  if (e < E) atomicAdd(&deg[dst[e]], 1.0f);
}

__global__ void rsqrt_kernel(float* __restrict__ dinv,
                             const float* __restrict__ deg, int n) {
  const int i = blockIdx.x * blockDim.x + threadIdx.x;
  if (i < n) dinv[i] = rsqrtf(deg[i]);
}

template<int DOUT>
__global__ void edge_scatter_kernel(const float* __restrict__ h,
                                    const int* __restrict__ src,
                                    const int* __restrict__ dst,
                                    const float* __restrict__ dinv,
                                    float* __restrict__ agg, int E) {
  const long long idx = (long long)blockIdx.x * blockDim.x + threadIdx.x;
  const int e = (int)(idx / DOUT);
  const int c = (int)(idx % DOUT);
  if (e >= E) return;
  const int s = src[e], d = dst[e];
  const float w = dinv[s] * dinv[d];
  atomicAdd(&agg[(size_t)d * DOUT + c], h[(size_t)s * DOUT + c] * w);
}

template<int DOUT>
__global__ void combine_relu_kernel(const float* __restrict__ agg,
                                    const float* __restrict__ h,
                                    const float* __restrict__ dinv,
                                    const float* __restrict__ bias,
                                    float* __restrict__ out, int n) {
  const long long idx = (long long)blockIdx.x * blockDim.x + threadIdx.x;
  const int i = (int)(idx / DOUT);
  const int c = (int)(idx % DOUT);
  if (i >= n) return;
  const float ns = dinv[i] * dinv[i];
  const float v = agg[(size_t)i * DOUT + c] + h[(size_t)i * DOUT + c] * ns + bias[c];
  out[(size_t)i * DOUT + c] = v > 0.0f ? v : 0.0f;
}

__global__ void classifier_kernel(const float* __restrict__ h,
                                  const float* __restrict__ Wc,
                                  const float* __restrict__ bc,
                                  float* __restrict__ out, int n) {
  const int i = blockIdx.x * blockDim.x + threadIdx.x;
  if (i >= n) return;
  float acc = bc[0];
  const float* row = h + (size_t)i * 64;
#pragma unroll
  for (int c = 0; c < 64; ++c) acc += row[c] * Wc[c];
  out[i] = acc;
}

// ---------------------------------------------------------------------------
// Host launcher
// ---------------------------------------------------------------------------
extern "C" void kernel_launch(void* const* d_in, const int* in_sizes, int n_in,
                              void* d_out, int out_size, void* d_ws, size_t ws_size,
                              hipStream_t stream) {
  const float* x  = (const float*)d_in[0];
  const int*   ei = (const int*)d_in[1];
  const float* W1 = (const float*)d_in[2];
  const float* b1 = (const float*)d_in[3];
  const float* W2 = (const float*)d_in[4];
  const float* b2 = (const float*)d_in[5];
  const float* W3 = (const float*)d_in[6];
  const float* b3 = (const float*)d_in[7];
  const float* Wc = (const float*)d_in[8];
  const float* bc = (const float*)d_in[9];

  const int DIN = 256, DH = 128, DH2 = 64;
  const int N = in_sizes[0] / DIN;
  const int E = in_sizes[1] / 2;
  const int* src = ei;
  const int* dst = ei + E;

  char* ws = (char*)d_ws;
  float* deg  = (float*)ws; ws += (size_t)N * 4;
  float* dinv = (float*)ws; ws += (size_t)N * 4;
  float* bufA = (float*)ws; ws += (size_t)N * DH * 4;   // GEMM output h
  float* bufB = (float*)ws; ws += (size_t)N * DH * 4;   // agg / layer out (odd)
  float* bufC = (float*)ws; ws += (size_t)N * DH * 4;   // agg / layer out (even)
  unsigned short* Hb = (unsigned short*)ws; ws += (size_t)N * DH * 2;  // bf16 activations
  unsigned short* Wt = (unsigned short*)ws; ws += (size_t)DIN * DH * 2; // bf16 W^T scratch
  // Alias: bf16 copy of x (N*256*2 B) shares memory with bufB (N*128*4 B, same
  // size). bufB's lifetime starts only after GEMM-1 has consumed Xb.
  unsigned short* Xb = (unsigned short*)bufB;

  auto cdiv = [](long long a, long long b) { return (unsigned)((a + b - 1) / b); };
  const unsigned MT = cdiv(N, 16);   // row tiles (N=100000 -> 6250, exact)

  // Degree / normalization.
  fill_kernel<<<cdiv(N, 256), 256, 0, stream>>>(deg, 1.0f, N);
  deg_accum_kernel<<<cdiv(E, 256), 256, 0, stream>>>(dst, deg, E);
  rsqrt_kernel<<<cdiv(N, 256), 256, 0, stream>>>(dinv, deg, N);

  // ---- Layer 1: [N,256] @ [256,128] ----
  cvt_f32_bf16_kernel<<<cdiv((long long)N * DIN / 4, 256), 256, 0, stream>>>(x, Xb, N * DIN / 4);
  cvt_w_tr_kernel<<<cdiv(DIN * DH, 256), 256, 0, stream>>>(W1, Wt, DIN, DH);
  gemm_bf16_wmma<256, 128><<<MT, 256, 0, stream>>>(Xb, Wt, bufA, N);
  fill_kernel<<<cdiv((long long)N * DH, 256), 256, 0, stream>>>(bufB, 0.0f, N * DH); // kills Xb
  edge_scatter_kernel<128><<<cdiv((long long)E * DH, 256), 256, 0, stream>>>(
      bufA, src, dst, dinv, bufB, E);
  combine_relu_kernel<128><<<cdiv((long long)N * DH, 256), 256, 0, stream>>>(
      bufB, bufA, dinv, b1, bufB, N);

  // ---- Layer 2: [N,128] @ [128,128] ----
  cvt_f32_bf16_kernel<<<cdiv((long long)N * DH / 4, 256), 256, 0, stream>>>(bufB, Hb, N * DH / 4);
  cvt_w_tr_kernel<<<cdiv(DH * DH, 256), 256, 0, stream>>>(W2, Wt, DH, DH);
  gemm_bf16_wmma<128, 128><<<MT, 256, 0, stream>>>(Hb, Wt, bufA, N);
  fill_kernel<<<cdiv((long long)N * DH, 256), 256, 0, stream>>>(bufC, 0.0f, N * DH);
  edge_scatter_kernel<128><<<cdiv((long long)E * DH, 256), 256, 0, stream>>>(
      bufA, src, dst, dinv, bufC, E);
  combine_relu_kernel<128><<<cdiv((long long)N * DH, 256), 256, 0, stream>>>(
      bufC, bufA, dinv, b2, bufC, N);

  // ---- Layer 3: [N,128] @ [128,64] ----
  cvt_f32_bf16_kernel<<<cdiv((long long)N * DH / 4, 256), 256, 0, stream>>>(bufC, Hb, N * DH / 4);
  cvt_w_tr_kernel<<<cdiv(DH * DH2, 256), 256, 0, stream>>>(W3, Wt, DH, DH2);
  gemm_bf16_wmma<128, 64><<<MT, 128, 0, stream>>>(Hb, Wt, bufA, N);
  fill_kernel<<<cdiv((long long)N * DH2, 256), 256, 0, stream>>>(bufB, 0.0f, N * DH2);
  edge_scatter_kernel<64><<<cdiv((long long)E * DH2, 256), 256, 0, stream>>>(
      bufA, src, dst, dinv, bufB, E);
  combine_relu_kernel<64><<<cdiv((long long)N * DH2, 256), 256, 0, stream>>>(
      bufB, bufA, dinv, b3, bufB, N);

  // ---- Classifier: [N,64] @ [64,1] + bc ----
  classifier_kernel<<<cdiv(N, 256), 256, 0, stream>>>(bufB, Wc, bc, (float*)d_out, N);
}
